// AngleNet_37280316130037
// MI455X (gfx1250) — matrix operand order repaired and checked by reading, not compiled, not run on hardware
//
#include <hip/hip_runtime.h>
#include <hip/hip_bf16.h>

// ---------------- problem constants ----------------
#define N_ATOMS 50000
#define A_ANG   200000
#define B_SEG   100
#define FR      256
#define DIN     512           // 2*FR
#define HDIM    256
#define P_ENS   6
#define THETA0_H 1.3824382542f  // sqrt(109.5*pi/180)
#define K_H      3.1622776601f  // sqrt(10)

// ---------------- tiling ----------------
#define MTILE 64              // angles per workgroup (4 row-tiles of 16)
#define NWAVE 8               // 256 threads
#define LDA_X (DIN + 8)       // padded bf16 row stride (conflict-free b128)
#define LDA_H (HDIM + 8)

// ---------------- workspace layout (dwords) ----------------
// [0..127]        : prefix sums of num_angles (101 ints)
// [128 ..]        : W1 packed  P * 16jt * 16ks * 256 dwords
// [.. ]           : W2 packed  P * 16jt *  8ks * 256 dwords
#define WS_PREFIX_DW 0
#define WS_W1_DW     128
#define W1_DW_P      (16 * 16 * 256)   // 65536 dwords per p
#define WS_W2_DW     (WS_W1_DW + P_ENS * W1_DW_P)
#define W2_DW_P      (16 * 8 * 256)    // 32768 dwords per p
// total ws: (128 + 6*65536 + 6*32768)*4 ~= 2.26 MB

typedef __attribute__((ext_vector_type(16))) __bf16 v16bf;
typedef __attribute__((ext_vector_type(8)))  float  v8f;

union FragAB { uint4 q[2]; v16bf v; };

__device__ __forceinline__ unsigned short f2bf(float f) {
  unsigned u = __builtin_bit_cast(unsigned, f);
  u += 0x7FFFu + ((u >> 16) & 1u);          // round-to-nearest-even
  return (unsigned short)(u >> 16);
}
__device__ __forceinline__ unsigned pk2bf(float lo, float hi) {
  return (unsigned)f2bf(lo) | ((unsigned)f2bf(hi) << 16);
}

// ---------------------------------------------------------------------------
// Prep 1: zero output, prefix-sum num_angles
// ---------------------------------------------------------------------------
__global__ void zero_prefix_kernel(const int* __restrict__ num_angles,
                                   float* __restrict__ out,
                                   int* __restrict__ prefix) {
  int t = threadIdx.x;
  if (t < B_SEG) out[t] = 0.0f;
  if (t == 0) {
    int s = 0;
    prefix[0] = 0;
    for (int i = 0; i < B_SEG; ++i) { s += num_angles[i]; prefix[i + 1] = s; }
  }
}

// ---------------------------------------------------------------------------
// Prep 2: pack W1/W2 (f32, [P][K][H] row-major) into bf16 B-fragment blocks.
// Block (p, jt, ks) = 32x16 K-panel: dword d -> lane=d/8, r=d%8,
//   nn = lane&15 (column), kh = lane>>4, k = ks*32 + kh*16 + 2r (+1 in hi)
// ---------------------------------------------------------------------------
__global__ void pack_weights_kernel(const float* __restrict__ W1,
                                    const float* __restrict__ W2,
                                    unsigned* __restrict__ ws) {
  int blk = blockIdx.x;
  int d   = threadIdx.x;                 // dword within 256-dword block
  int lane = d >> 3, r = d & 7;
  int nn = lane & 15, kh = lane >> 4;
  if (blk < P_ENS * 256) {               // ---- W1: 256 blocks per p ----
    int p = blk >> 8, rem = blk & 255;   // rem = jt*16 + ks
    int jt = rem >> 4, ks = rem & 15;
    int k  = ks * 32 + kh * 16 + 2 * r;
    int col = jt * 16 + nn;
    const float* w = W1 + ((size_t)p * DIN + k) * HDIM + col;
    ws[WS_W1_DW + (size_t)p * W1_DW_P + (size_t)rem * 256 + d] =
        pk2bf(w[0], w[HDIM]);
  } else {                               // ---- W2: 128 blocks per p ----
    int b = blk - P_ENS * 256;
    int p = b >> 7, rem = b & 127;       // rem = jt*8 + ks
    int jt = rem >> 3, ks = rem & 7;
    int k  = ks * 32 + kh * 16 + 2 * r;
    int col = jt * 16 + nn;
    const float* w = W2 + ((size_t)p * HDIM + k) * HDIM + col;
    ws[WS_W2_DW + (size_t)p * W2_DW_P + (size_t)rem * 256 + d] =
        pk2bf(w[0], w[HDIM]);
  }
}

// ---------------------------------------------------------------------------
// Main fused kernel
// ---------------------------------------------------------------------------
struct __align__(16) SMem {
  unsigned short X [MTILE * LDA_X];      // 66560 B : node_input tile (bf16)
  unsigned short H1[MTILE * LDA_H];      // 33792 B
  unsigned short H2[MTILE * LDA_H];      // 33792 B
  float theta[MTILE];
  float outp[P_ENS][MTILE];
  float scratch[4][MTILE];
  int   prefix[B_SEG + 4];
};                                        // ~137 KB (<320 KB/WGP)

// One MLP layer: D = tanh(A(16x..K) x Wpack + bias), A from LDS, out -> LDS.
// Each of 8 waves owns 2 column tiles (jt = wave*2 + j) x 4 row tiles.
template <int KSTEPS>
__device__ __forceinline__ void mlp_layer(const unsigned short* __restrict__ ldsA,
                                          int lda,
                                          const uint4* __restrict__ wpack,
                                          const float* __restrict__ bias,
                                          unsigned short* __restrict__ ldsOut,
                                          int ldo) {
  const int tid  = threadIdx.x;
  const int wave = tid >> 5;
  const int lane = tid & 31;
  const int m    = lane & 15;            // A row / D column-in-tile
  const int kh   = lane >> 4;            // K-half select

  v8f acc[4][2];
  #pragma unroll
  for (int rt = 0; rt < 4; ++rt)
    #pragma unroll
    for (int j = 0; j < 2; ++j) acc[rt][j] = v8f{};

  for (int ks = 0; ks < KSTEPS; ++ks) {
    FragAB a[4];
    #pragma unroll
    for (int rt = 0; rt < 4; ++rt) {     // A frag: 2x ds_load_b128
      const unsigned short* ap =
          ldsA + (size_t)(rt * 16 + m) * lda + ks * 32 + kh * 16;
      a[rt].q[0] = *(const uint4*)(ap);
      a[rt].q[1] = *(const uint4*)(ap + 8);
    }
    #pragma unroll
    for (int j = 0; j < 2; ++j) {
      const int jt = wave * 2 + j;
      const uint4* bp = wpack + (size_t)(jt * KSTEPS + ks) * 64 + lane * 2;
      FragAB b;
      b.q[0] = bp[0];                    // 2x global_load_b128, coalesced
      b.q[1] = bp[1];
      __builtin_prefetch(bp + 64, 0, 1); // next ks panel -> global_prefetch
      #pragma unroll
      for (int rt = 0; rt < 4; ++rt)
        acc[rt][j] = __builtin_amdgcn_wmma_f32_16x16x32_bf16(
            false, a[rt].v, false, b.v, (short)0, acc[rt][j], false, false);
    }
  }

  // epilogue: bias + tanh, store bf16 to LDS (D layout: m = v + 8*kh, n = m-lane)
  #pragma unroll
  for (int j = 0; j < 2; ++j) {
    const int jt  = wave * 2 + j;
    const int col = jt * 16 + m;
    const float bv = bias[col];
    #pragma unroll
    for (int rt = 0; rt < 4; ++rt) {
      #pragma unroll
      for (int v = 0; v < 8; ++v) {
        float val = tanhf(acc[rt][j][v] + bv);
        int row = rt * 16 + v + kh * 8;
        ldsOut[(size_t)row * ldo + col] = f2bf(val);
      }
    }
  }
}

__global__ __launch_bounds__(256) void anglenet_kernel(
    const float* __restrict__ r, const float* __restrict__ xyz,
    const int* __restrict__ angles,
    const float* __restrict__ b1, const float* __restrict__ b2,
    const float* __restrict__ W3, const float* __restrict__ b3,
    const unsigned* __restrict__ ws, float* __restrict__ out) {
  __shared__ SMem s;
  const int tid  = threadIdx.x;
  const int base = blockIdx.x * MTILE;   // A_ANG % MTILE == 0 -> no tail

  if (tid <= B_SEG) s.prefix[tid] = ((const int*)ws)[WS_PREFIX_DW + tid];

  // ---- phase 0: gather node_input -> bf16 X tile; theta per angle ----
  {
    const int row = tid >> 2;            // 4 threads per row
    const int c0  = (tid & 3) * 128;     // each covers 128 columns
    const int g   = base + row;
    const int a0 = angles[3 * g], a1 = angles[3 * g + 1], a2 = angles[3 * g + 2];
    if (c0 < FR) {                       // cols [0,256): r[a0] + r[a2]
      const float* ra = r + (size_t)a0 * FR + c0;
      const float* rc = r + (size_t)a2 * FR + c0;
      for (int c = 0; c < 128; c += 4) {
        float4 x = *(const float4*)(ra + c);
        float4 y = *(const float4*)(rc + c);
        *(uint2*)&s.X[(size_t)row * LDA_X + c0 + c] =
            make_uint2(pk2bf(x.x + y.x, x.y + y.y), pk2bf(x.z + y.z, x.w + y.w));
      }
    } else {                             // cols [256,512): r[a1]
      const float* rb = r + (size_t)a1 * FR + (c0 - FR);
      for (int c = 0; c < 128; c += 4) {
        float4 x = *(const float4*)(rb + c);
        *(uint2*)&s.X[(size_t)row * LDA_X + c0 + c] =
            make_uint2(pk2bf(x.x, x.y), pk2bf(x.z, x.w));
      }
    }
    if (tid < MTILE) {
      const int gg = base + tid;
      const int i0 = angles[3 * gg], i1 = angles[3 * gg + 1], i2 = angles[3 * gg + 2];
      float v1x = xyz[3 * i1] - xyz[3 * i0], v1y = xyz[3 * i1 + 1] - xyz[3 * i0 + 1],
            v1z = xyz[3 * i1 + 2] - xyz[3 * i0 + 2];
      float v2x = xyz[3 * i2] - xyz[3 * i1], v2y = xyz[3 * i2 + 1] - xyz[3 * i1 + 1],
            v2z = xyz[3 * i2 + 2] - xyz[3 * i1 + 2];
      float dot = -(v1x * v2x + v1y * v2y + v1z * v2z);
      float nrm = sqrtf((v1x * v1x + v1y * v1y + v1z * v1z) *
                        (v2x * v2x + v2y * v2y + v2z * v2z));
      float ct = (dot / nrm) * (1.0f / 1.000001f);
      s.theta[tid] = acosf(ct);
    }
  }
  __syncthreads();

  const uint4* w1base = (const uint4*)(ws + WS_W1_DW);
  const uint4* w2base = (const uint4*)(ws + WS_W2_DW);

  // ---- per-ensemble MLP, fully on-chip ----
  for (int p = 0; p < P_ENS; ++p) {
    mlp_layer<16>(s.X, LDA_X, w1base + (size_t)p * (W1_DW_P / 4),
                  b1 + p * HDIM, s.H1, LDA_H);
    __syncthreads();
    mlp_layer<8>(s.H1, LDA_H, w2base + (size_t)p * (W2_DW_P / 4),
                 b2 + p * HDIM, s.H2, LDA_H);
    __syncthreads();
    {                                    // GEMV: out = H2 . W3[p] + b3[p]
      const int a = tid & 63, part = tid >> 6;   // 4 partials per angle
      const float* w3 = W3 + p * HDIM + part * 64;
      const unsigned* h2 =
          (const unsigned*)&s.H2[(size_t)a * LDA_H + part * 64];
      float sum = 0.f;
      #pragma unroll 8
      for (int k = 0; k < 32; ++k) {     // 32 bf16-pairs
        unsigned u = h2[k];
        float lo = __builtin_bit_cast(float, u << 16);
        float hi = __builtin_bit_cast(float, u & 0xFFFF0000u);
        sum += lo * w3[2 * k] + hi * w3[2 * k + 1];
      }
      s.scratch[part][a] = sum;
    }
    __syncthreads();
    if (tid < MTILE)
      s.outp[p][tid] = s.scratch[0][tid] + s.scratch[1][tid] +
                       s.scratch[2][tid] + s.scratch[3][tid] + b3[p];
    __syncthreads();
  }

  // ---- energy polynomial + segment sum ----
  if (tid < MTILE) {
    const int g = base + tid;
    const float th = s.theta[tid];
    float o0 = s.outp[0][tid], o1 = s.outp[1][tid], o2 = s.outp[2][tid],
          o3 = s.outp[3][tid], o4 = s.outp[4][tid], o5 = s.outp[5][tid];
    float t0h = (THETA0_H + o0); t0h *= t0h;
    float kh  = (K_H + o1);      kh  *= kh;
    float dh = th - t0h;
    float E = 0.5f * kh * dh * dh;
    float dc = th - o2 * o2;
    E += 0.5f * (o3 * o3) * dc * dc * dc;
    float dq = th - o4 * o4;
    float dq2 = dq * dq;
    E += 0.5f * (o5 * o5) * dq2 * dq2;
    int lo = 0, hi = B_SEG;              // bucket: prefix[b] <= g < prefix[b+1]
    while (lo < hi) {
      int mid = (lo + hi) >> 1;
      if (s.prefix[mid + 1] <= g) lo = mid + 1; else hi = mid;
    }
    atomicAdd(&out[lo], E);
  }
}

// ---------------------------------------------------------------------------
extern "C" void kernel_launch(void* const* d_in, const int* in_sizes, int n_in,
                              void* d_out, int out_size, void* d_ws,
                              size_t ws_size, hipStream_t stream) {
  const float* r          = (const float*)d_in[0];
  const float* xyz        = (const float*)d_in[1];
  const int*   angles     = (const int*)d_in[2];
  const int*   num_angles = (const int*)d_in[3];
  const float* W1         = (const float*)d_in[4];
  const float* b1         = (const float*)d_in[5];
  const float* W2         = (const float*)d_in[6];
  const float* b2         = (const float*)d_in[7];
  const float* W3         = (const float*)d_in[8];
  const float* b3         = (const float*)d_in[9];
  float*    out = (float*)d_out;
  unsigned* ws  = (unsigned*)d_ws;   // needs ~2.3 MB

  zero_prefix_kernel<<<1, 128, 0, stream>>>(num_angles, out,
                                            (int*)(ws + WS_PREFIX_DW));
  pack_weights_kernel<<<P_ENS * 256 + P_ENS * 128, 256, 0, stream>>>(W1, W2, ws);
  anglenet_kernel<<<A_ANG / MTILE, 256, 0, stream>>>(r, xyz, angles, b1, b2,
                                                     W3, b3, ws, out);
}
